// MultiheadAttention_5085241279148
// MI455X (gfx1250) — matrix operand order repaired and checked by reading, not compiled, not run on hardware
//
#include <hip/hip_runtime.h>

#define DEV static __device__ __forceinline__

typedef _Float16 h16 __attribute__((ext_vector_type(16)));
typedef _Float16 h8  __attribute__((ext_vector_type(8)));
typedef _Float16 h4  __attribute__((ext_vector_type(4)));
typedef float    f8  __attribute__((ext_vector_type(8)));

constexpr int T_LEN = 2048;
constexpr int BSZ_  = 2;
constexpr int EMB   = 1024;
constexpr int NH    = 16;
constexpr int HD    = 64;
constexpr int BH    = 32;                 // BSZ_*NH
constexpr int NTOK  = T_LEN * BSZ_;       // 4096
constexpr int E3    = 3 * EMB;            // 3072
constexpr float SCALE_Q = 0.125f;         // HD^-0.5
constexpr float LOG2E   = 1.44269504088896340736f;

// ---------- fragment helpers (contiguous k order k = 16*(lane>>4)+e in both A and B) ----------

DEV h16 cat8(h8 lo, h8 hi) {
  return __builtin_shufflevector(lo, hi, 0,1,2,3,4,5,6,7,8,9,10,11,12,13,14,15);
}

// A fragment 16x32 f16: lane holds row M=lane&15; elements e = K(base+e),
// base = (lane>>4)*16.  rowp points at K=0.
DEV h16 loadA32(const _Float16* rowp, int abase) {
  h8 lo = *(const h8*)(rowp + abase);
  h8 hi = *(const h8*)(rowp + abase + 8);
  return cat8(lo, hi);
}

// B fragment 32x16 f16: lane holds column N=lane&15; elements e = B[kb+e][N],
// kb=(lane>>4)*16.  With the "N" matrix stored N-major (K contiguous), this is
// 16 contiguous halves; p points at that run.
DEV h16 loadB32(const _Float16* p) {
  h8 lo = *(const h8*)(p);
  h8 hi = *(const h8*)(p + 8);
  return cat8(lo, hi);
}

DEV f8 zero8() {
  f8 r;
#pragma unroll
  for (int i = 0; i < 8; ++i) r[i] = 0.0f;
  return r;
}

DEV f8 wmma16(h16 a, h16 b, f8 c) {
  return __builtin_amdgcn_wmma_f32_16x16x32_f16(false, a, false, b, (short)0, c,
                                                false, false);
}

// reductions across the 16-lane half (C-fragment rows are constant per half)
DEV float rmax16(float v) {
  v = fmaxf(v, __shfl_xor(v, 1, 32));
  v = fmaxf(v, __shfl_xor(v, 2, 32));
  v = fmaxf(v, __shfl_xor(v, 4, 32));
  v = fmaxf(v, __shfl_xor(v, 8, 32));
  return v;
}
DEV float rsum16(float v) {
  v += __shfl_xor(v, 1, 32);
  v += __shfl_xor(v, 2, 32);
  v += __shfl_xor(v, 4, 32);
  v += __shfl_xor(v, 8, 32);
  return v;
}

// ---------------------------- kernels ----------------------------

// vectorized f32 -> f16 conversion, 4 elements per thread
__global__ __launch_bounds__(256) void k_cvt_f16(const float4* __restrict__ in,
                                                 h4* __restrict__ out, int n4) {
  int i = blockIdx.x * 256 + threadIdx.x;
  if (i < n4) {
    float4 v = in[i];
    h4 o;
    o[0] = (_Float16)v.x; o[1] = (_Float16)v.y;
    o[2] = (_Float16)v.z; o[3] = (_Float16)v.w;
    out[i] = o;
  }
}

// QKV projection: [NTOK,EMB] x [EMB,E3] (B = Wt, W row-major so K contiguous).
// Ping-pong double-buffered pipeline: fragment set loads for chunk c+1 issue
// before the WMMAs of chunk c, with no register-rotation moves.
// Epilogue scatters into Q[bh][t][d] (scaled), K[bh][t][d], Vt[bh][d][t].
__global__ __launch_bounds__(256) void k_qkv(const _Float16* __restrict__ X,
                                             const _Float16* __restrict__ W,
                                             const float* __restrict__ bias,
                                             _Float16* __restrict__ Q,
                                             _Float16* __restrict__ K,
                                             _Float16* __restrict__ Vt) {
  const int lane = threadIdx.x & 31, wid = threadIdx.x >> 5;
  const int task = blockIdx.x * 8 + wid;
  const int mt = task / (E3 / 64), nt = task % (E3 / 64);
  const int row0 = mt * 16, n0 = nt * 64;
  const int M = lane & 15, lh = lane >> 4;
  const int abase = lh * 16, kb = lh * 16;

  f8 acc[4];
#pragma unroll
  for (int j = 0; j < 4; ++j) acc[j] = zero8();

  const _Float16* arow = X + (size_t)(row0 + M) * EMB;
  const _Float16* brow[4];
#pragma unroll
  for (int j = 0; j < 4; ++j)
    brow[j] = W + (size_t)(n0 + j * 16 + M) * EMB + kb;

  auto ldB = [&](h16 (&b)[4], int kk) {
#pragma unroll
    for (int j = 0; j < 4; ++j) b[j] = loadB32(brow[j] + kk);
  };
  auto mm4 = [&](h16 a, h16 (&b)[4]) {
#pragma unroll
    for (int j = 0; j < 4; ++j) acc[j] = wmma16(a, b[j], acc[j]);
  };

  h16 a0 = loadA32(arow, abase), a1;
  h16 b0[4], b1[4];
  ldB(b0, 0);
  for (int kk = 32; kk < EMB - 64; kk += 64) {
    a1 = loadA32(arow + kk, abase);      ldB(b1, kk);
    mm4(a0, b0);
    a0 = loadA32(arow + kk + 32, abase); ldB(b0, kk + 32);
    mm4(a1, b1);
  }
  a1 = loadA32(arow + (EMB - 32), abase); ldB(b1, EMB - 32);
  mm4(a0, b0);
  mm4(a1, b1);

#pragma unroll
  for (int j = 0; j < 4; ++j) {
    const int n = n0 + j * 16 + M;
    const float bv = bias[n];
#pragma unroll
    for (int i = 0; i < 8; ++i) {
      const int row = row0 + i + 8 * lh;       // token index t*BSZ+b
      const int t = row >> 1, bb = row & 1;
      const float v = acc[j][i] + bv;
      if (n < EMB) {
        int h = n >> 6, d = n & 63;
        Q[(((size_t)(bb * NH + h)) * T_LEN + t) * HD + d] = (_Float16)(v * SCALE_Q);
      } else if (n < 2 * EMB) {
        int nn = n - EMB, h = nn >> 6, d = nn & 63;
        K[(((size_t)(bb * NH + h)) * T_LEN + t) * HD + d] = (_Float16)v;
      } else {
        int nn = n - 2 * EMB, h = nn >> 6, d = nn & 63;
        Vt[(((size_t)(bb * NH + h)) * HD + d) * T_LEN + t] = (_Float16)v;
      }
    }
  }
}

// Flash attention: one wave = one (head, 16-query tile). Online base-2 softmax.
// V fragments load up-front each k-chunk so they overlap QK WMMAs and the
// softmax VALU block.  Emits context (f16) and per-row stats m/l.
__global__ __launch_bounds__(256) void k_flash(const _Float16* __restrict__ Q,
                                               const _Float16* __restrict__ K,
                                               const _Float16* __restrict__ Vt,
                                               _Float16* __restrict__ Ctx,
                                               float* __restrict__ Sm,
                                               float* __restrict__ Sl) {
  __shared__ _Float16 plds[8][16 * 32];
  const int lane = threadIdx.x & 31, wid = threadIdx.x >> 5;
  const int task = blockIdx.x * 8 + wid;
  const int bh = task >> 7, qt = task & 127;
  const int q0 = qt * 16;
  const int M = lane & 15, lh = lane >> 4;
  const int abase = lh * 16, kb = lh * 16;
  const int b = bh >> 4, h = bh & 15;

  const _Float16* qrow = Q + ((size_t)bh * T_LEN + q0 + M) * HD;
  h16 aq0 = loadA32(qrow, abase);        // d 0..31
  h16 aq1 = loadA32(qrow + 32, abase);   // d 32..63

  float m[8], l[8];
#pragma unroll
  for (int i = 0; i < 8; ++i) { m[i] = -1e30f; l[i] = 0.0f; }
  f8 acc[4];
#pragma unroll
  for (int j = 0; j < 4; ++j) acc[j] = zero8();

  for (int k0 = 0; k0 < T_LEN; k0 += 32) {
    // ---- issue ALL loads for this chunk first (K then V) ----
    const _Float16* krow0 = K + ((size_t)bh * T_LEN + k0 + M) * HD + kb;
    const _Float16* krow1 = krow0 + 16 * HD;
    h16 bk00 = loadB32(krow0);
    h16 bk01 = loadB32(krow0 + 32);
    h16 bk10 = loadB32(krow1);
    h16 bk11 = loadB32(krow1 + 32);
    h16 bv[4];
#pragma unroll
    for (int dj = 0; dj < 4; ++dj)
      bv[dj] = loadB32(Vt + ((size_t)bh * HD + dj * 16 + M) * T_LEN + k0 + kb);

    f8 s0 = zero8(), s1 = zero8();
    s0 = wmma16(aq0, bk00, s0);
    s0 = wmma16(aq1, bk01, s0);
    s1 = wmma16(aq0, bk10, s1);
    s1 = wmma16(aq1, bk11, s1);

#pragma unroll
    for (int i = 0; i < 8; ++i) {
      float a0 = s0[i] * LOG2E, a1 = s1[i] * LOG2E;
      float mc = rmax16(fmaxf(a0, a1));
      float mn = fmaxf(m[i], mc);
      float sc = __builtin_amdgcn_exp2f(m[i] - mn);
      float p0 = __builtin_amdgcn_exp2f(a0 - mn);
      float p1 = __builtin_amdgcn_exp2f(a1 - mn);
      l[i] = l[i] * sc + rsum16(p0 + p1);
      m[i] = mn;
      acc[0][i] *= sc; acc[1][i] *= sc; acc[2][i] *= sc; acc[3][i] *= sc;
      plds[wid][(i + 8 * lh) * 32 + M]      = (_Float16)p0;
      plds[wid][(i + 8 * lh) * 32 + 16 + M] = (_Float16)p1;
    }
    asm volatile("s_wait_dscnt 0x0" ::: "memory");
    h8 plo = *(const h8*)&plds[wid][M * 32 + abase];
    h8 phi = *(const h8*)&plds[wid][M * 32 + abase + 8];
    h16 ap = cat8(plo, phi);
#pragma unroll
    for (int dj = 0; dj < 4; ++dj)
      acc[dj] = wmma16(ap, bv[dj], acc[dj]);
  }

#pragma unroll
  for (int i = 0; i < 8; ++i) {
    const float inv = 1.0f / l[i];
    const int q = q0 + i + 8 * lh;
#pragma unroll
    for (int dj = 0; dj < 4; ++dj)
      Ctx[((size_t)q * BSZ_ + b) * EMB + h * HD + dj * 16 + M] =
          (_Float16)(acc[dj][i] * inv);
    if (M == 0) {
      Sm[(size_t)bh * T_LEN + q] = m[i];
      Sl[(size_t)bh * T_LEN + q] = l[i];
    }
  }
}

// Output projection: Ctx [NTOK,EMB] x out_w^T -> f32 attn output [T,B,E].
// Same ping-pong pipeline as k_qkv.
__global__ __launch_bounds__(256) void k_outproj(const _Float16* __restrict__ Ctx,
                                                 const _Float16* __restrict__ W,
                                                 const float* __restrict__ bias,
                                                 float* __restrict__ Out) {
  const int lane = threadIdx.x & 31, wid = threadIdx.x >> 5;
  const int task = blockIdx.x * 8 + wid;
  const int nt = task & 15, mt = task >> 4;
  const int row0 = mt * 16, n0 = nt * 64;
  const int M = lane & 15, lh = lane >> 4;
  const int abase = lh * 16, kb = lh * 16;

  f8 acc[4];
#pragma unroll
  for (int j = 0; j < 4; ++j) acc[j] = zero8();

  const _Float16* arow = Ctx + (size_t)(row0 + M) * EMB;
  const _Float16* brow[4];
#pragma unroll
  for (int j = 0; j < 4; ++j)
    brow[j] = W + (size_t)(n0 + j * 16 + M) * EMB + kb;

  auto ldB = [&](h16 (&b)[4], int kk) {
#pragma unroll
    for (int j = 0; j < 4; ++j) b[j] = loadB32(brow[j] + kk);
  };
  auto mm4 = [&](h16 a, h16 (&b)[4]) {
#pragma unroll
    for (int j = 0; j < 4; ++j) acc[j] = wmma16(a, b[j], acc[j]);
  };

  h16 a0 = loadA32(arow, abase), a1;
  h16 b0[4], b1[4];
  ldB(b0, 0);
  for (int kk = 32; kk < EMB - 64; kk += 64) {
    a1 = loadA32(arow + kk, abase);      ldB(b1, kk);
    mm4(a0, b0);
    a0 = loadA32(arow + kk + 32, abase); ldB(b0, kk + 32);
    mm4(a1, b1);
  }
  a1 = loadA32(arow + (EMB - 32), abase); ldB(b1, EMB - 32);
  mm4(a0, b0);
  mm4(a1, b1);

#pragma unroll
  for (int j = 0; j < 4; ++j) {
    const int n = n0 + j * 16 + M;
    const float bv = bias[n];
#pragma unroll
    for (int i = 0; i < 8; ++i)
      Out[(size_t)(row0 + i + 8 * lh) * EMB + n] = acc[j][i] + bv;
  }
}

// Head-averaged probabilities: recompute S tiles via WMMA, normalize with the
// stored (m, l) stats, average over 16 heads, write [B][T][T] f32.
// Ping-pong over heads: next head's 6 fragments load before current WMMAs+exp.
__global__ __launch_bounds__(256) void k_avgw(const _Float16* __restrict__ Q,
                                              const _Float16* __restrict__ K,
                                              const float* __restrict__ Sm,
                                              const float* __restrict__ Sl,
                                              float* __restrict__ Out) {
  const int lane = threadIdx.x & 31, wid = threadIdx.x >> 5;
  const int task = blockIdx.x * 8 + wid;
  const int kc = task & 63, qt = (task >> 6) & 127, b = task >> 13;
  const int q0 = qt * 16, k0 = kc * 32;
  const int M = lane & 15, lh = lane >> 4;
  const int abase = lh * 16, kb = lh * 16;

  f8 acc0 = zero8(), acc1 = zero8();

  auto ldSet = [&](int h, h16& aq0, h16& aq1, h16 (&bk)[4]) {
    const int bh = b * NH + h;
    const _Float16* qr = Q + ((size_t)bh * T_LEN + q0 + M) * HD;
    const _Float16* kr = K + ((size_t)bh * T_LEN + k0 + M) * HD + kb;
    aq0 = loadA32(qr, abase);
    aq1 = loadA32(qr + 32, abase);
    bk[0] = loadB32(kr);
    bk[1] = loadB32(kr + 32);
    bk[2] = loadB32(kr + 16 * HD);
    bk[3] = loadB32(kr + 16 * HD + 32);
  };
  auto compute = [&](int h, h16 aq0, h16 aq1, h16 (&bk)[4]) {
    const int bh = b * NH + h;
    f8 s0 = zero8(), s1 = zero8();
    s0 = wmma16(aq0, bk[0], s0);
    s0 = wmma16(aq1, bk[1], s0);
    s1 = wmma16(aq0, bk[2], s1);
    s1 = wmma16(aq1, bk[3], s1);
#pragma unroll
    for (int i = 0; i < 8; ++i) {
      const int q = q0 + i + 8 * lh;
      const float mi = Sm[(size_t)bh * T_LEN + q];
      const float wi = 1.0f / (Sl[(size_t)bh * T_LEN + q] * (float)NH);
      acc0[i] += __builtin_amdgcn_exp2f(s0[i] * LOG2E - mi) * wi;
      acc1[i] += __builtin_amdgcn_exp2f(s1[i] * LOG2E - mi) * wi;
    }
  };

  h16 aq0a, aq1a, aq0b, aq1b;
  h16 bka[4], bkb[4];
  ldSet(0, aq0a, aq1a, bka);
  for (int h = 0; h < NH - 2; h += 2) {
    ldSet(h + 1, aq0b, aq1b, bkb);
    compute(h, aq0a, aq1a, bka);
    ldSet(h + 2, aq0a, aq1a, bka);
    compute(h + 1, aq0b, aq1b, bkb);
  }
  ldSet(NH - 1, aq0b, aq1b, bkb);
  compute(NH - 2, aq0a, aq1a, bka);
  compute(NH - 1, aq0b, aq1b, bkb);

#pragma unroll
  for (int i = 0; i < 8; ++i) {
    const int q = q0 + i + 8 * lh;
    const size_t base = ((size_t)b * T_LEN + q) * T_LEN + k0;
    Out[base + M]      = acc0[i];
    Out[base + 16 + M] = acc1[i];
  }
}

// ---------------------------- launch ----------------------------

extern "C" void kernel_launch(void* const* d_in, const int* in_sizes, int n_in,
                              void* d_out, int out_size, void* d_ws, size_t ws_size,
                              hipStream_t stream) {
  const float* q_in  = (const float*)d_in[0];   // [T,B,E]
  const float* w_in  = (const float*)d_in[1];   // [3E,E]
  const float* b_in  = (const float*)d_in[2];   // [3E]
  const float* ow_in = (const float*)d_in[3];   // [E,E]
  const float* ob_in = (const float*)d_in[4];   // [E]

  char* ws = (char*)d_ws;
  size_t off = 0;
  auto take = [&](size_t bytes) { char* p = ws + off; off += bytes; return p; };

  _Float16* Xbf  = (_Float16*)take((size_t)NTOK * EMB * 2);
  _Float16* Wbf  = (_Float16*)take((size_t)E3 * EMB * 2);
  _Float16* OWbf = (_Float16*)take((size_t)EMB * EMB * 2);
  _Float16* Qws  = (_Float16*)take((size_t)BH * T_LEN * HD * 2);
  _Float16* Kws  = (_Float16*)take((size_t)BH * T_LEN * HD * 2);
  _Float16* Vtws = (_Float16*)take((size_t)BH * T_LEN * HD * 2);
  _Float16* Ctx  = (_Float16*)take((size_t)NTOK * EMB * 2);
  float*    Sm   = (float*)take((size_t)BH * T_LEN * 4);
  float*    Sl   = (float*)take((size_t)BH * T_LEN * 4);

  float* attn_out = (float*)d_out;                       // [T,B,E]
  float* avg_out  = attn_out + (size_t)NTOK * EMB;       // [B,T,T]

  k_cvt_f16<<<NTOK * EMB / 1024, 256, 0, stream>>>((const float4*)q_in,
                                                   (h4*)Xbf, NTOK * EMB / 4);
  k_cvt_f16<<<E3 * EMB / 1024, 256, 0, stream>>>((const float4*)w_in,
                                                 (h4*)Wbf, E3 * EMB / 4);
  k_cvt_f16<<<EMB * EMB / 1024, 256, 0, stream>>>((const float4*)ow_in,
                                                  (h4*)OWbf, EMB * EMB / 4);

  k_qkv<<<(NTOK / 16) * (E3 / 64) / 8, 256, 0, stream>>>(Xbf, Wbf, b_in,
                                                         Qws, Kws, Vtws);
  k_flash<<<BH * (T_LEN / 16) / 8, 256, 0, stream>>>(Qws, Kws, Vtws, Ctx, Sm, Sl);
  k_outproj<<<(NTOK / 16) * (EMB / 64) / 8, 256, 0, stream>>>(Ctx, OWbf, ob_in,
                                                              attn_out);
  k_avgw<<<BSZ_ * (T_LEN / 16) * (T_LEN / 32) / 8, 256, 0, stream>>>(Qws, Kws,
                                                                     Sm, Sl,
                                                                     avg_out);
}
